// GroupGAT_ICP_47708496724516
// MI455X (gfx1250) — compile-verified, matrix operand order checked by reading, not run on hardware
//
#include <hip/hip_runtime.h>
#include <cmath>

// ---------------------------------------------------------------------------
// GroupGAT forward for MI455X (gfx1250, wave32).
// Dense GEMMs run on v_wmma_f32_16x16x32_bf16 (bf16 in, f32 accumulate).
// ---------------------------------------------------------------------------

typedef __attribute__((ext_vector_type(16))) __bf16 v16bf;
typedef __attribute__((ext_vector_type(8)))  float  v8f;

#define NUM_MOLS 4096
#define N_ATOM   131072
#define E_ATOMC  262144
#define N_FRAG   32768
#define E_JT     65536

__device__ __forceinline__ unsigned short f2bf(float f) {
    unsigned u = __float_as_uint(f);
    unsigned r = u + 0x7FFFu + ((u >> 16) & 1u);   // round-to-nearest-even
    return (unsigned short)(r >> 16);
}

// K offset pattern of the 16-bit WMMA A fragment (16x32), per CDNA5 ISA 7.12.2.
__device__ __forceinline__ int wmma_kbase(int v, int half) {
    return (v < 4) ? (2 * v + 8 * half) : (16 + 2 * (v - 4) + 8 * half);
}

// ------------------------- weight packing (f32 -> bf16 B-fragments) --------
// W[K,N] row-major f32 -> tiles (kt,nt), 32 lanes x 16 contiguous bf16/lane.
// B layout mirrors the A layout with n <-> m (lane = n%16 + 16*half).
__global__ void pack_w_kernel(const float* __restrict__ W,
                              unsigned short* __restrict__ out, int K, int N) {
    int KT = (K + 31) >> 5, NT = (N + 15) >> 4;
    int t = blockIdx.x * blockDim.x + threadIdx.x;
    if (t >= KT * NT * 32) return;
    int lane = t & 31, tile = t >> 5;
    int nt = tile % NT, kt = tile / NT;
    int half = lane >> 4;
    int n = nt * 16 + (lane & 15);
    unsigned short* o = out + (size_t)t * 16;
#pragma unroll
    for (int e = 0; e < 16; ++e) {
        int v = e >> 1;
        int k = kt * 32 + wmma_kbase(v, half) + (e & 1);
        float val = (k < K && n < N) ? W[(size_t)k * N + n] : 0.0f;
        o[e] = f2bf(val);
    }
}

// ------------------------------ WMMA GEMM ----------------------------------
// C[M,N](ldc) = act(Arow @ W + bias),  Arow = concat(A1[idx? idx[r]:r], A2[r])
// 128 threads = 4 waves; 64-row M tile; wave w: rows w*16..w*16+15, NT n-tiles.
template <int NT, int ACT>
__global__ __launch_bounds__(128) void gemm_kernel(
    const float* __restrict__ A1, const int* __restrict__ idx,
    const float* __restrict__ A2, int K1, int K2,
    const unsigned short* __restrict__ Wp, const float* __restrict__ bias,
    float* __restrict__ C, int ldc, int M, int N, int NTtot) {
    __shared__ unsigned short smA[64 * 384];   // 48 KB, Kpad <= 384
    const int K = K1 + K2;
    const int KT = (K + 31) >> 5;
    const int Kpad = KT << 5;
    const int m0 = blockIdx.x * 64;
    const int ntOff = blockIdx.y * NT;
    const int tid = threadIdx.x;

    // cooperative A-tile stage: f32 -> bf16, zero-padded
    for (int i = tid; i < 64 * Kpad; i += 128) {
        int row = i / Kpad, k = i - row * Kpad;
        float val = 0.0f;
        int gr = m0 + row;
        if (gr < M) {
            if (k < K1) {
                int r = idx ? idx[gr] : gr;
                val = A1[(size_t)r * K1 + k];
            } else if (k < K) {
                val = A2[(size_t)gr * K2 + (k - K1)];
            }
        }
        smA[i] = f2bf(val);
    }
    __syncthreads();

    const int lane = tid & 31, wave = tid >> 5;
    const int m = lane & 15, half = lane >> 4;

    v8f acc[NT];
    const v8f vzero = {0.f, 0.f, 0.f, 0.f, 0.f, 0.f, 0.f, 0.f};
#pragma unroll
    for (int i = 0; i < NT; ++i) acc[i] = vzero;

    const unsigned short* rowbase = &smA[(wave * 16 + m) * Kpad];
    for (int kt = 0; kt < KT; ++kt) {
        union { unsigned u[8]; v16bf v; } a;
        const unsigned short* rp = rowbase + (kt << 5);
#pragma unroll
        for (int v = 0; v < 8; ++v)
            a.u[v] = *(const unsigned*)(rp + wmma_kbase(v, half));
#pragma unroll
        for (int nt = 0; nt < NT; ++nt) {
            const v16bf b = *(const v16bf*)(Wp +
                ((size_t)(kt * NTtot + ntOff + nt) * 32 + lane) * 16);
            acc[nt] = __builtin_amdgcn_wmma_f32_16x16x32_bf16(
                false, a.v, false, b, (short)0, acc[nt], false, false);
        }
    }
    // epilogue: C layout VGPR j -> row j + 8*half, col = lane&15
#pragma unroll
    for (int nt = 0; nt < NT; ++nt) {
        int col = (ntOff + nt) * 16 + (lane & 15);
        if (col >= N) continue;
        float bb = bias ? bias[col] : 0.0f;
#pragma unroll
        for (int j = 0; j < 8; ++j) {
            int row = m0 + wave * 16 + j + 8 * half;
            if (row < M) {
                float x = acc[nt][j] + bb;
                if (ACT == 1) x = x > 0.f ? x : 0.01f * x;
                else if (ACT == 2) x = x > 0.f ? x : 0.f;
                else if (ACT == 3) x = x > 0.f ? x : 0.001f * x;
                C[(size_t)row * ldc + col] = x;
            }
        }
    }
}

// ------------------------------ misc kernels -------------------------------
__global__ void fill_kernel(float* __restrict__ p, float v, size_t n) {
    size_t i = (size_t)blockIdx.x * blockDim.x + threadIdx.x;
    if (i < n) p[i] = v;
}
__global__ void relu_kernel(float* __restrict__ p, size_t n) {
    size_t i = (size_t)blockIdx.x * blockDim.x + threadIdx.x;
    if (i < n) p[i] = fmaxf(p[i], 0.0f);
}

// sc[e] = leaky( Xs[sidx?sidx[e]:e] . attL + Xd[didx[e]] . attR , 0.01 )
__global__ void edge_score_kernel(const float* __restrict__ Xs, const int* __restrict__ sidx,
                                  const float* __restrict__ attL,
                                  const float* __restrict__ Xd, const int* __restrict__ didx,
                                  const float* __restrict__ attR,
                                  float* __restrict__ sc, int E) {
    int wv = (blockIdx.x * blockDim.x + threadIdx.x) >> 5;
    int lane = threadIdx.x & 31;
    if (wv >= E) return;
    int rs = sidx ? sidx[wv] : wv;
    int rd = didx[wv];
    const float* xs = Xs + (size_t)rs * 128;
    const float* xd = Xd + (size_t)rd * 128;
    float p = 0.f;
#pragma unroll
    for (int i = 0; i < 4; ++i) {
        int c = lane + 32 * i;
        p += xs[c] * attL[c] + xd[c] * attR[c];
    }
#pragma unroll
    for (int off = 16; off > 0; off >>= 1) p += __shfl_xor(p, off, 32);
    if (lane == 0) sc[wv] = p > 0.f ? p : 0.01f * p;
}

__global__ void seg_max_kernel(const float* __restrict__ sc, const int* __restrict__ seg,
                               float* __restrict__ mx, int E) {
    int i = blockIdx.x * blockDim.x + threadIdx.x;
    if (i >= E) return;
    float v = sc[i];
    float* p = mx + seg[i];
    if (v >= 0.f) atomicMax((int*)p, __float_as_int(v));
    else          atomicMin((unsigned int*)p, __float_as_uint(v));
}
__global__ void exp_sum_kernel(const float* __restrict__ sc, const int* __restrict__ seg,
                               const float* __restrict__ mx, float* __restrict__ den,
                               float* __restrict__ eexp, int E) {
    int i = blockIdx.x * blockDim.x + threadIdx.x;
    if (i >= E) return;
    int d = seg[i];
    float e = expf(sc[i] - mx[d]);
    eexp[i] = e;
    atomicAdd(den + d, e);
}
__global__ void norm_kernel(const float* __restrict__ eexp, const int* __restrict__ seg,
                            const float* __restrict__ den, float* __restrict__ a, int E) {
    int i = blockIdx.x * blockDim.x + threadIdx.x;
    if (i >= E) return;
    a[i] = eexp[i] / (den[seg[i]] + 1e-16f);
}

// out[seg[e], c] += (w?w[e]:1) * X[gidx?gidx[e]:e, c]   (D=128)
__global__ void scatter_wsum_kernel(const float* __restrict__ w, const float* __restrict__ X,
                                    const int* __restrict__ gidx, const int* __restrict__ seg,
                                    float* __restrict__ out, int E) {
    size_t t = (size_t)blockIdx.x * blockDim.x + threadIdx.x;
    if (t >= (size_t)E * 128) return;
    int e = (int)(t >> 7), c = (int)(t & 127);
    int r = gidx ? gidx[e] : e;
    float v = X[(size_t)r * 128 + c];
    if (w) v *= w[e];
    atomicAdd(out + (size_t)seg[e] * 128 + c, v);
}

// hout = elu(gru_combine(gi[M,384], gh[M,384], h[M,128]))
__global__ void gru_combine_kernel(const float* __restrict__ gi, const float* __restrict__ gh,
                                   const float* __restrict__ h, float* __restrict__ out, int M) {
    size_t t = (size_t)blockIdx.x * blockDim.x + threadIdx.x;
    if (t >= (size_t)M * 128) return;
    int r = (int)(t >> 7), c = (int)(t & 127);
    size_t b = (size_t)r * 384 + c;
    float ir = gi[b], iz = gi[b + 128], in_ = gi[b + 256];
    float hr = gh[b], hz = gh[b + 128], hn = gh[b + 256];
    float rr = 1.f / (1.f + expf(-(ir + hr)));
    float zz = 1.f / (1.f + expf(-(iz + hz)));
    float nn = tanhf(in_ + rr * hn);
    float ho = (1.f - zz) * nn + zz * h[t];
    out[t] = ho > 0.f ? ho : expf(ho) - 1.f;   // ELU
}

// sc[n] = leaky( concat(g[batch[n]], h[n]) . w[256] + b , 0.01 )
__global__ void pool_score_kernel(const float* __restrict__ g, const int* __restrict__ batch,
                                  const float* __restrict__ h, const float* __restrict__ w,
                                  const float* __restrict__ b, float* __restrict__ sc, int Nn) {
    int wv = (blockIdx.x * blockDim.x + threadIdx.x) >> 5;
    int lane = threadIdx.x & 31;
    if (wv >= Nn) return;
    const float* gp = g + (size_t)batch[wv] * 128;
    const float* hp = h + (size_t)wv * 128;
    float p = 0.f;
#pragma unroll
    for (int i = 0; i < 4; ++i) {
        int c = lane + 32 * i;
        p += gp[c] * w[c] + hp[c] * w[128 + c];
    }
#pragma unroll
    for (int off = 16; off > 0; off >>= 1) p += __shfl_xor(p, off, 32);
    if (lane == 0) {
        float v = p + b[0];
        sc[wv] = v > 0.f ? v : 0.01f * v;
    }
}

__global__ void mean_relu_kernel(const float* __restrict__ hj, float* __restrict__ out, int G) {
    size_t i = (size_t)blockIdx.x * blockDim.x + threadIdx.x;
    if (i >= (size_t)G * 128) return;
    int r = (int)(i >> 7), c = (int)(i & 127);
    float v = 0.5f * (hj[(size_t)r * 256 + c] + hj[(size_t)r * 256 + 128 + c]);
    out[i] = fmaxf(v, 0.0f);
}
__global__ void concat3_kernel(const float* __restrict__ a, const float* __restrict__ b,
                               const float* __restrict__ c, float* __restrict__ out, int G) {
    size_t i = (size_t)blockIdx.x * blockDim.x + threadIdx.x;
    if (i >= (size_t)G * 384) return;
    int r = (int)(i / 384), col = (int)(i % 384);
    float v;
    if (col < 128) v = a[(size_t)r * 128 + col];
    else if (col < 256) v = b[(size_t)r * 128 + col - 128];
    else v = c[(size_t)r * 128 + col - 256];
    out[i] = v;
}
__global__ void final_kernel(const float* __restrict__ op, const float* __restrict__ gfeat,
                             float* __restrict__ out, int G) {
    int i = blockIdx.x * blockDim.x + threadIdx.x;
    if (i >= G) return;
    float B = op[i * 5 + 0], Cc = op[i * 5 + 1], D = op[i * 5 + 2];
    float E = op[i * 5 + 3], F = op[i * 5 + 4];
    float T = gfeat[i] + 1e-7f;
    float DT = fminf(fmaxf(D / T, -1000.f), 1000.f);
    float FT = fminf(fmaxf(F / T, -1000.f), 1000.f);
    float r1 = DT / sinhf(DT);
    float r2 = FT / coshf(FT);
    out[i] = B + Cc * r1 * r1 + E * r2 * r2;
}

// ------------------------------ host side ----------------------------------
static inline unsigned cdiv_sz(size_t a, size_t b) { return (unsigned)((a + b - 1) / b); }
static inline void* ws_alloc(char*& p, size_t bytes) {
    void* r = (void*)p;
    p += (bytes + 255) & ~(size_t)255;
    return r;
}

enum { ACT_NONE = 0, ACT_LRELU = 1, ACT_RELU = 2, ACT_LRELU3 = 3 };

static void run_gemm(hipStream_t st, const float* A1, const int* idx, const float* A2,
                     int K1, int K2, const unsigned short* Wp, const float* bias,
                     float* C, int ldc, int M, int N, int act) {
    int NTtot = (N + 15) / 16;
    dim3 blk(128);
    unsigned gx = (unsigned)((M + 63) / 64);
    if (NTtot % 8 == 0) {
        dim3 grd(gx, (unsigned)(NTtot / 8));
        switch (act) {
        case ACT_NONE:  gemm_kernel<8, 0><<<grd, blk, 0, st>>>(A1, idx, A2, K1, K2, Wp, bias, C, ldc, M, N, NTtot); break;
        case ACT_LRELU: gemm_kernel<8, 1><<<grd, blk, 0, st>>>(A1, idx, A2, K1, K2, Wp, bias, C, ldc, M, N, NTtot); break;
        case ACT_RELU:  gemm_kernel<8, 2><<<grd, blk, 0, st>>>(A1, idx, A2, K1, K2, Wp, bias, C, ldc, M, N, NTtot); break;
        default:        gemm_kernel<8, 3><<<grd, blk, 0, st>>>(A1, idx, A2, K1, K2, Wp, bias, C, ldc, M, N, NTtot); break;
        }
    } else if (NTtot == 12) {
        dim3 grd(gx, 1);
        switch (act) {
        case ACT_NONE:  gemm_kernel<12, 0><<<grd, blk, 0, st>>>(A1, idx, A2, K1, K2, Wp, bias, C, ldc, M, N, NTtot); break;
        default:        gemm_kernel<12, 3><<<grd, blk, 0, st>>>(A1, idx, A2, K1, K2, Wp, bias, C, ldc, M, N, NTtot); break;
        }
    } else {
        dim3 grd(gx, 1);
        gemm_kernel<1, 0><<<grd, blk, 0, st>>>(A1, idx, A2, K1, K2, Wp, bias, C, ldc, M, N, NTtot);
    }
}

static unsigned short* pack_weight(const float* W, int K, int N, char*& wsp, hipStream_t st) {
    int KT = (K + 31) / 32, NT = (N + 15) / 16;
    size_t cnt = (size_t)KT * NT * 512;
    unsigned short* o = (unsigned short*)ws_alloc(wsp, cnt * sizeof(unsigned short));
    int total = KT * NT * 32;
    pack_w_kernel<<<cdiv_sz((size_t)total, 128), 128, 0, st>>>(W, o, K, N);
    return o;
}

// params arrive as jax tree_leaves (sorted-dict) order; AFP leaf block:
enum {
    AFP_ATT_L = 0, AFP_ATT_R, AFP_G_ATT_DST, AFP_G_ATT_SRC,
    AFP_G_GRU_BH, AFP_G_GRU_BI, AFP_G_GRU_WH, AFP_G_GRU_WI,
    AFP_G_W_B, AFP_G_W_W,
    AFP_GRU0_BH, AFP_GRU0_BI, AFP_GRU0_WH, AFP_GRU0_WI,
    AFP_GRUM_BH, AFP_GRUM_BI, AFP_GRUM_WH, AFP_GRUM_WI,
    AFP_LIN1_B, AFP_LIN1_W, AFP_LIN2_B, AFP_LIN2_W,
    AFP_NODE_B, AFP_NODE_W, AFP_OUT_B, AFP_OUT_W,
    AFP_PA_B, AFP_PA_W, AFP_PL_B, AFP_PL_W, AFP_NLEAF
};
// top-level leaf bases (after the 14 data inputs)
enum {
    PB = 14,
    I_FRAG_AFP = PB,                 // 30 leaves
    I_FRAG_ATT_B = PB + 30, I_FRAG_ATT_W,
    I_JT_EDGE_B, I_JT_EDGE_W, I_JT_FRAG_B, I_JT_FRAG_W,
    I_JT_AFP = PB + 36,              // 30 leaves
    I_JT_PROJ_B = PB + 66, I_JT_PROJ_W,
    I_ORIG_AFP = PB + 68,            // 30 leaves
    I_ORIG_ATT_B = PB + 98, I_ORIG_ATT_W,
    I_ORIG_EDGE_B, I_ORIG_EDGE_W, I_ORIG_NODE_B, I_ORIG_NODE_W,
    I_PRED1_B, I_PRED1_W, I_PRED2A_B, I_PRED2A_W, I_PRED2B_B, I_PRED2B_W,
    I_TOTAL
};

struct AfpW {
    const float *att_l, *att_r, *g_att_dst, *g_att_src;
    const float *g_gru_bh, *g_gru_bi, *gru0_bh, *gru0_bi, *grum_bh, *grum_bi;
    unsigned short *g_gru_wh, *g_gru_wi, *gru0_wh, *gru0_wi, *grum_wh, *grum_wi;
    const float* g_w_b;  unsigned short* g_w_w;
    const float* lin1_b; unsigned short* lin1_w; int lin1_K;
    const float* lin2_b; unsigned short* lin2_w;
    const float* node_b; unsigned short* node_w; int node_K;
    const float* out_b;  unsigned short* out_w;
    const float *pa_w, *pa_b;
    const float* pl_b;   unsigned short* pl_w;
};

static AfpW make_afp(void* const* d_in, int base, int h, int node_in, int edge_in,
                     char*& wsp, hipStream_t st) {
    auto F = [&](int leaf) -> const float* { return (const float*)d_in[base + leaf]; };
    AfpW w{};
    w.att_l = F(AFP_ATT_L) + (size_t)h * 128;
    w.att_r = F(AFP_ATT_R) + (size_t)h * 128;
    w.g_att_dst = F(AFP_G_ATT_DST) + (size_t)h * 128;
    w.g_att_src = F(AFP_G_ATT_SRC) + (size_t)h * 128;
    w.g_gru_bh = F(AFP_G_GRU_BH) + (size_t)h * 384;
    w.g_gru_bi = F(AFP_G_GRU_BI) + (size_t)h * 384;
    w.g_gru_wh = pack_weight(F(AFP_G_GRU_WH) + (size_t)h * 128 * 384, 128, 384, wsp, st);
    w.g_gru_wi = pack_weight(F(AFP_G_GRU_WI) + (size_t)h * 128 * 384, 128, 384, wsp, st);
    w.g_w_b = F(AFP_G_W_B) + (size_t)h * 128;
    w.g_w_w = pack_weight(F(AFP_G_W_W) + (size_t)h * 128 * 128, 128, 128, wsp, st);
    w.gru0_bh = F(AFP_GRU0_BH) + (size_t)h * 384;
    w.gru0_bi = F(AFP_GRU0_BI) + (size_t)h * 384;
    w.gru0_wh = pack_weight(F(AFP_GRU0_WH) + (size_t)h * 128 * 384, 128, 384, wsp, st);
    w.gru0_wi = pack_weight(F(AFP_GRU0_WI) + (size_t)h * 128 * 384, 128, 384, wsp, st);
    w.grum_bh = F(AFP_GRUM_BH) + (size_t)h * 384;
    w.grum_bi = F(AFP_GRUM_BI) + (size_t)h * 384;
    w.grum_wh = pack_weight(F(AFP_GRUM_WH) + (size_t)h * 128 * 384, 128, 384, wsp, st);
    w.grum_wi = pack_weight(F(AFP_GRUM_WI) + (size_t)h * 128 * 384, 128, 384, wsp, st);
    w.lin1_K = 128 + edge_in;
    w.lin1_b = F(AFP_LIN1_B) + (size_t)h * 128;
    w.lin1_w = pack_weight(F(AFP_LIN1_W) + (size_t)h * w.lin1_K * 128, w.lin1_K, 128, wsp, st);
    w.lin2_b = F(AFP_LIN2_B) + (size_t)h * 128;
    w.lin2_w = pack_weight(F(AFP_LIN2_W) + (size_t)h * 128 * 128, 128, 128, wsp, st);
    w.node_K = node_in;
    w.node_b = F(AFP_NODE_B) + (size_t)h * 128;
    w.node_w = pack_weight(F(AFP_NODE_W) + (size_t)h * node_in * 128, node_in, 128, wsp, st);
    w.out_b = F(AFP_OUT_B) + (size_t)h * 128;
    w.out_w = pack_weight(F(AFP_OUT_W) + (size_t)h * 128 * 128, 128, 128, wsp, st);
    w.pa_b = F(AFP_PA_B) + h;
    w.pa_w = F(AFP_PA_W) + (size_t)h * 256;
    w.pl_b = F(AFP_PL_B) + (size_t)h * 128;
    w.pl_w = pack_weight(F(AFP_PL_W) + (size_t)h * 128 * 128, 128, 128, wsp, st);
    return w;
}

struct Scratch {
    float *h, *hb, *z, *t, *gi, *gh, *msg, *g, *gb;
    float *sc, *eexp, *aw, *nmax, *nden;
};

static void seg_softmax(hipStream_t st, const Scratch& S, const int* seg, int E, int nseg) {
    fill_kernel<<<cdiv_sz((size_t)nseg, 256), 256, 0, st>>>(S.nmax, -INFINITY, (size_t)nseg);
    fill_kernel<<<cdiv_sz((size_t)nseg, 256), 256, 0, st>>>(S.nden, 0.0f, (size_t)nseg);
    seg_max_kernel<<<cdiv_sz((size_t)E, 256), 256, 0, st>>>(S.sc, seg, S.nmax, E);
    exp_sum_kernel<<<cdiv_sz((size_t)E, 256), 256, 0, st>>>(S.sc, seg, S.nmax, S.nden, S.eexp, E);
    norm_kernel<<<cdiv_sz((size_t)E, 256), 256, 0, st>>>(S.eexp, seg, S.nden, S.aw, E);
}

static void run_afp(hipStream_t st, const AfpW& w, const float* x, int node_in,
                    const int* src, const int* dst, const float* eattr, int edge_in,
                    const int* batch, int Nn, int E, int G,
                    float* Cout, int ldc, const Scratch& S) {
    // h = lrelu(x @ node)
    run_gemm(st, x, nullptr, nullptr, node_in, 0, w.node_w, w.node_b, S.h, 128, Nn, 128, ACT_LRELU);
    float* h = S.h; float* hb = S.hb;
    // ej = lrelu(concat(h[src], eattr) @ lin1)
    run_gemm(st, h, src, eattr, 128, edge_in, w.lin1_w, w.lin1_b, S.t, 128, E, 128, ACT_LRELU);
    edge_score_kernel<<<cdiv_sz((size_t)E * 32, 256), 256, 0, st>>>(S.t, nullptr, w.att_l, h, dst, w.att_r, S.sc, E);
    seg_softmax(st, S, dst, E, Nn);
    // t = h[src] @ lin2 + b ; msg = seg_sum(a * t, dst)
    run_gemm(st, h, src, nullptr, 128, 0, w.lin2_w, w.lin2_b, S.t, 128, E, 128, ACT_NONE);
    fill_kernel<<<cdiv_sz((size_t)Nn * 128, 256), 256, 0, st>>>(S.msg, 0.0f, (size_t)Nn * 128);
    scatter_wsum_kernel<<<cdiv_sz((size_t)E * 128, 256), 256, 0, st>>>(S.aw, S.t, nullptr, dst, S.msg, E);
    // h = elu(gru0(msg, h))
    run_gemm(st, S.msg, nullptr, nullptr, 128, 0, w.gru0_wi, w.gru0_bi, S.gi, 384, Nn, 384, ACT_NONE);
    run_gemm(st, h, nullptr, nullptr, 128, 0, w.gru0_wh, w.gru0_bh, S.gh, 384, Nn, 384, ACT_NONE);
    gru_combine_kernel<<<cdiv_sz((size_t)Nn * 128, 256), 256, 0, st>>>(S.gi, S.gh, h, hb, Nn);
    { float* tmp = h; h = hb; hb = tmp; }
    // GAT layer
    run_gemm(st, h, nullptr, nullptr, 128, 0, w.g_w_w, w.g_w_b, S.z, 128, Nn, 128, ACT_NONE);
    edge_score_kernel<<<cdiv_sz((size_t)E * 32, 256), 256, 0, st>>>(S.z, src, w.g_att_src, S.z, dst, w.g_att_dst, S.sc, E);
    seg_softmax(st, S, dst, E, Nn);
    fill_kernel<<<cdiv_sz((size_t)Nn * 128, 256), 256, 0, st>>>(S.msg, 0.0f, (size_t)Nn * 128);
    scatter_wsum_kernel<<<cdiv_sz((size_t)E * 128, 256), 256, 0, st>>>(S.aw, S.z, src, dst, S.msg, E);
    run_gemm(st, S.msg, nullptr, nullptr, 128, 0, w.g_gru_wi, w.g_gru_bi, S.gi, 384, Nn, 384, ACT_NONE);
    run_gemm(st, h, nullptr, nullptr, 128, 0, w.g_gru_wh, w.g_gru_bh, S.gh, 384, Nn, 384, ACT_NONE);
    gru_combine_kernel<<<cdiv_sz((size_t)Nn * 128, 256), 256, 0, st>>>(S.gi, S.gh, h, hb, Nn);
    { float* tmp = h; h = hb; hb = tmp; }
    // mol pooling:  g = relu(seg_sum(h, batch))
    fill_kernel<<<cdiv_sz((size_t)G * 128, 256), 256, 0, st>>>(S.g, 0.0f, (size_t)G * 128);
    scatter_wsum_kernel<<<cdiv_sz((size_t)Nn * 128, 256), 256, 0, st>>>(nullptr, h, nullptr, batch, S.g, Nn);
    relu_kernel<<<cdiv_sz((size_t)G * 128, 256), 256, 0, st>>>(S.g, (size_t)G * 128);
    // pl = h @ pool_lin + b (loop-invariant) -> reuse S.z
    run_gemm(st, h, nullptr, nullptr, 128, 0, w.pl_w, w.pl_b, S.z, 128, Nn, 128, ACT_NONE);
    float* g = S.g; float* gb = S.gb;
    for (int it = 0; it < 2; ++it) {
        pool_score_kernel<<<cdiv_sz((size_t)Nn * 32, 256), 256, 0, st>>>(g, batch, h, w.pa_w, w.pa_b, S.sc, Nn);
        seg_softmax(st, S, batch, Nn, G);
        fill_kernel<<<cdiv_sz((size_t)G * 128, 256), 256, 0, st>>>(S.msg, 0.0f, (size_t)G * 128);
        scatter_wsum_kernel<<<cdiv_sz((size_t)Nn * 128, 256), 256, 0, st>>>(S.aw, S.z, nullptr, batch, S.msg, Nn);
        run_gemm(st, S.msg, nullptr, nullptr, 128, 0, w.grum_wi, w.grum_bi, S.gi, 384, G, 384, ACT_NONE);
        run_gemm(st, g, nullptr, nullptr, 128, 0, w.grum_wh, w.grum_bh, S.gh, 384, G, 384, ACT_NONE);
        gru_combine_kernel<<<cdiv_sz((size_t)G * 128, 256), 256, 0, st>>>(S.gi, S.gh, g, gb, G);
        { float* tmp = g; g = gb; gb = tmp; }
    }
    // head output: g @ out + b -> Cout (ldc)
    run_gemm(st, g, nullptr, nullptr, 128, 0, w.out_w, w.out_b, Cout, ldc, G, 128, ACT_NONE);
}

extern "C" void kernel_launch(void* const* d_in, const int* in_sizes, int n_in,
                              void* d_out, int out_size, void* d_ws, size_t ws_size,
                              hipStream_t stream) {
    (void)in_sizes; (void)out_size; (void)ws_size;
    if (n_in < I_TOTAL) return;

    const float* data_x  = (const float*)d_in[0];
    const int*   d_src   = (const int*)d_in[1];
    const int*   d_dst   = d_src + E_ATOMC;
    const float* data_ea = (const float*)d_in[2];
    const int*   d_batch = (const int*)d_in[3];
    const float* frag_x  = (const float*)d_in[4];
    const int*   f_src   = (const int*)d_in[5];
    const int*   f_dst   = f_src + E_ATOMC;
    const float* frag_ea = (const float*)d_in[6];
    const int*   f_batch = (const int*)d_in[7];
    const float* junc_x  = (const float*)d_in[8];
    const int*   j_src   = (const int*)d_in[9];
    const int*   j_dst   = j_src + E_JT;
    const float* junc_ea = (const float*)d_in[10];
    const int*   j_batch = (const int*)d_in[11];
    const float* gfeat   = (const float*)d_in[13];
    float* out = (float*)d_out;

    char* wsp = (char*)d_ws;
    hipStream_t st = stream;

    // -------- pack all weights into WMMA-fragment bf16 layout --------------
    AfpW frag_w[2], jt_w[2], orig_w[2];
    for (int h = 0; h < 2; ++h) frag_w[h] = make_afp(d_in, I_FRAG_AFP, h, 44, 12, wsp, st);
    for (int h = 0; h < 2; ++h) jt_w[h]   = make_afp(d_in, I_JT_AFP,   h, 128, 128, wsp, st);
    for (int h = 0; h < 2; ++h) orig_w[h] = make_afp(d_in, I_ORIG_AFP, h, 128, 128, wsp, st);
    unsigned short* W_onode = pack_weight((const float*)d_in[I_ORIG_NODE_W], 44, 128, wsp, st);
    unsigned short* W_oedge = pack_weight((const float*)d_in[I_ORIG_EDGE_W], 12, 128, wsp, st);
    unsigned short* W_oatt  = pack_weight((const float*)d_in[I_ORIG_ATT_W], 256, 128, wsp, st);
    unsigned short* W_fatt  = pack_weight((const float*)d_in[I_FRAG_ATT_W], 256, 128, wsp, st);
    unsigned short* W_jfrag = pack_weight((const float*)d_in[I_JT_FRAG_W], 219, 128, wsp, st);
    unsigned short* W_jedge = pack_weight((const float*)d_in[I_JT_EDGE_W], 12, 128, wsp, st);
    unsigned short* W_proj[2];
    for (int h = 0; h < 2; ++h)
        W_proj[h] = pack_weight((const float*)d_in[I_JT_PROJ_W] + (size_t)h * 256 * 128, 256, 128, wsp, st);
    unsigned short* W_p1  = pack_weight((const float*)d_in[I_PRED1_W], 384, 192, wsp, st);
    unsigned short* W_p2a = pack_weight((const float*)d_in[I_PRED2A_W], 192, 192, wsp, st);
    unsigned short* W_p2b = pack_weight((const float*)d_in[I_PRED2B_W], 192, 5, wsp, st);

    // -------- persistent buffers -------------------------------------------
    float* EX = (float*)ws_alloc(wsp, (size_t)N_ATOM * 128 * 4);
    float* EE = (float*)ws_alloc(wsp, (size_t)E_ATOMC * 128 * 4);
    float* HO = (float*)ws_alloc(wsp, (size_t)NUM_MOLS * 256 * 4);
    float* GO = (float*)ws_alloc(wsp, (size_t)NUM_MOLS * 128 * 4);
    float* HF = (float*)ws_alloc(wsp, (size_t)N_FRAG * 256 * 4);
    float* GF = (float*)ws_alloc(wsp, (size_t)N_FRAG * 128 * 4);
    float* FR = (float*)ws_alloc(wsp, (size_t)NUM_MOLS * 128 * 4);
    float* ME = (float*)ws_alloc(wsp, (size_t)N_FRAG * 128 * 4);
    float* JE = (float*)ws_alloc(wsp, (size_t)E_JT * 128 * 4);
    float* XP = (float*)ws_alloc(wsp, (size_t)N_FRAG * 128 * 4);
    float* HJ = (float*)ws_alloc(wsp, (size_t)NUM_MOLS * 256 * 4);
    float* SN = (float*)ws_alloc(wsp, (size_t)NUM_MOLS * 128 * 4);
    float* FE = (float*)ws_alloc(wsp, (size_t)NUM_MOLS * 384 * 4);
    float* D1 = (float*)ws_alloc(wsp, (size_t)NUM_MOLS * 192 * 4);
    float* D2 = (float*)ws_alloc(wsp, (size_t)NUM_MOLS * 192 * 4);
    float* OP = (float*)ws_alloc(wsp, (size_t)NUM_MOLS * 5 * 4);

    // -------- reused AFP scratch (sized for the largest graphs) ------------
    Scratch S;
    S.h    = (float*)ws_alloc(wsp, (size_t)N_ATOM * 128 * 4);
    S.hb   = (float*)ws_alloc(wsp, (size_t)N_ATOM * 128 * 4);
    S.z    = (float*)ws_alloc(wsp, (size_t)N_ATOM * 128 * 4);
    S.msg  = (float*)ws_alloc(wsp, (size_t)N_ATOM * 128 * 4);
    S.t    = (float*)ws_alloc(wsp, (size_t)E_ATOMC * 128 * 4);
    S.gi   = (float*)ws_alloc(wsp, (size_t)N_ATOM * 384 * 4);
    S.gh   = (float*)ws_alloc(wsp, (size_t)N_ATOM * 384 * 4);
    S.g    = (float*)ws_alloc(wsp, (size_t)N_FRAG * 128 * 4);
    S.gb   = (float*)ws_alloc(wsp, (size_t)N_FRAG * 128 * 4);
    S.sc   = (float*)ws_alloc(wsp, (size_t)E_ATOMC * 4);
    S.eexp = (float*)ws_alloc(wsp, (size_t)E_ATOMC * 4);
    S.aw   = (float*)ws_alloc(wsp, (size_t)E_ATOMC * 4);
    S.nmax = (float*)ws_alloc(wsp, (size_t)N_ATOM * 4);
    S.nden = (float*)ws_alloc(wsp, (size_t)N_ATOM * 4);

    // -------- origin branch -------------------------------------------------
    run_gemm(st, data_x, nullptr, nullptr, 44, 0, W_onode, (const float*)d_in[I_ORIG_NODE_B], EX, 128, N_ATOM, 128, ACT_LRELU);
    run_gemm(st, data_ea, nullptr, nullptr, 12, 0, W_oedge, (const float*)d_in[I_ORIG_EDGE_B], EE, 128, E_ATOMC, 128, ACT_LRELU);
    for (int h = 0; h < 2; ++h)
        run_afp(st, orig_w[h], EX, 128, d_src, d_dst, EE, 128, d_batch,
                N_ATOM, E_ATOMC, NUM_MOLS, HO + (size_t)h * 128, 256, S);
    run_gemm(st, HO, nullptr, nullptr, 256, 0, W_oatt, (const float*)d_in[I_ORIG_ATT_B], GO, 128, NUM_MOLS, 128, ACT_RELU);

    // -------- fragment branch ----------------------------------------------
    for (int h = 0; h < 2; ++h)
        run_afp(st, frag_w[h], frag_x, 44, f_src, f_dst, frag_ea, 12, f_batch,
                N_ATOM, E_ATOMC, N_FRAG, HF + (size_t)h * 128, 256, S);
    run_gemm(st, HF, nullptr, nullptr, 256, 0, W_fatt, (const float*)d_in[I_FRAG_ATT_B], GF, 128, N_FRAG, 128, ACT_RELU);
    fill_kernel<<<cdiv_sz((size_t)NUM_MOLS * 128, 256), 256, 0, st>>>(FR, 0.0f, (size_t)NUM_MOLS * 128);
    scatter_wsum_kernel<<<cdiv_sz((size_t)N_FRAG * 128, 256), 256, 0, st>>>(nullptr, GF, nullptr, j_batch, FR, N_FRAG);

    // -------- junction-tree branch -----------------------------------------
    run_gemm(st, junc_x, nullptr, nullptr, 219, 0, W_jfrag, (const float*)d_in[I_JT_FRAG_B], ME, 128, N_FRAG, 128, ACT_LRELU);
    run_gemm(st, junc_ea, nullptr, nullptr, 12, 0, W_jedge, (const float*)d_in[I_JT_EDGE_B], JE, 128, E_JT, 128, ACT_LRELU);
    for (int h = 0; h < 2; ++h) {
        run_gemm(st, GF, nullptr, ME, 128, 128, W_proj[h],
                 (const float*)d_in[I_JT_PROJ_B] + (size_t)h * 128, XP, 128, N_FRAG, 128, ACT_NONE);
        run_afp(st, jt_w[h], XP, 128, j_src, j_dst, JE, 128, j_batch,
                N_FRAG, E_JT, NUM_MOLS, HJ + (size_t)h * 128, 256, S);
    }
    mean_relu_kernel<<<cdiv_sz((size_t)NUM_MOLS * 128, 256), 256, 0, st>>>(HJ, SN, NUM_MOLS);

    // -------- prediction head ----------------------------------------------
    concat3_kernel<<<cdiv_sz((size_t)NUM_MOLS * 384, 256), 256, 0, st>>>(GO, FR, SN, FE, NUM_MOLS);
    run_gemm(st, FE, nullptr, nullptr, 384, 0, W_p1, (const float*)d_in[I_PRED1_B], D1, 192, NUM_MOLS, 192, ACT_LRELU3);
    run_gemm(st, D1, nullptr, nullptr, 192, 0, W_p2a, (const float*)d_in[I_PRED2A_B], D2, 192, NUM_MOLS, 192, ACT_LRELU3);
    run_gemm(st, D2, nullptr, nullptr, 192, 0, W_p2b, (const float*)d_in[I_PRED2B_B], OP, 5, NUM_MOLS, 5, ACT_NONE);
    final_kernel<<<cdiv_sz((size_t)NUM_MOLS, 256), 256, 0, st>>>(OP, gfeat, out, NUM_MOLS);
}
